// swin_attn_block_3D_89661737271836
// MI455X (gfx1250) — matrix operand order repaired and checked by reading, not compiled, not run on hardware
//
#include <hip/hip_runtime.h>

// ---------------------------------------------------------------------------
// CDNA5 (gfx1250, wave32) fused 3-branch Swin window-attention.
// One 256-thread workgroup (8 wave32) per 8x8 window; all GEMMs via
// v_wmma_f32_16x16x32_f16 with f32 accumulation; LDS staging between stages.
// ---------------------------------------------------------------------------

typedef __attribute__((ext_vector_type(16))) _Float16 v16h;
typedef __attribute__((ext_vector_type(8)))  _Float16 v8h;
typedef __attribute__((ext_vector_type(8)))  float    v8f;

#define N_TOK 64     // tokens per 8x8 window
#define DIM_  96     // channels
#define HD_   32     // head dim
#define XO_STR 104   // half-stride for X / O staging (16B-aligned rows)
#define QK_STR 104
#define VT_STR 72
#define S_STR  66
#define P_STR  72

// A-fragment (16x32 f16, row-major source). Layout: lanes 0-15 row M=lane,
// K 0-15; lanes 16-31 same rows, K 16-31 (contiguous, matches frag_bt).
// -> two 16-byte contiguous loads per lane.
__device__ __forceinline__ v16h frag_a(const _Float16* p, int stride, int r0,
                                       int k0, int lane) {
  const _Float16* a = p + (r0 + (lane & 15)) * stride + k0 + ((lane >> 4) << 4);
  v8h lo = *(const v8h*)(a);
  v8h hi = *(const v8h*)(a + 8);
  v16h f;
#pragma unroll
  for (int i = 0; i < 8; ++i) { f[i] = lo[i]; f[i + 8] = hi[i]; }
  return f;
}

// B-fragment (32x16 f16) from TRANSPOSED storage [n][k] (i.e. row-major K / W):
// lane n = lane&15, K base (lane>>4)*16 -> 16 contiguous halves.
__device__ __forceinline__ v16h frag_bt(const _Float16* p, int stride, int n0,
                                        int k0, int lane) {
  const _Float16* a = p + (n0 + (lane & 15)) * stride + k0 + ((lane >> 4) << 4);
  v8h lo = *(const v8h*)(a);
  v8h hi = *(const v8h*)(a + 8);
  v16h f;
#pragma unroll
  for (int i = 0; i < 8; ++i) { f[i] = lo[i]; f[i + 8] = hi[i]; }
  return f;
}

#define WMMA(acc, A, B)                                                        \
  acc = __builtin_amdgcn_wmma_f32_16x16x32_f16(false, (A), false, (B),         \
                                               (short)0, (acc), false, false)

// ---------------------------------------------------------------------------
// Prep: weights fp32->f16 (q-scale folded into Q rows), RPB bias table gather,
// BN fold, scaled qkv bias copy.
// ---------------------------------------------------------------------------
__global__ __launch_bounds__(256) void swin3d_prep_kernel(
    const float* __restrict__ qkv_w, const float* __restrict__ qkv_b,
    const float* __restrict__ proj_w, const float* __restrict__ rpb_table,
    const int* __restrict__ rpb_index, const float* __restrict__ bn_g,
    const float* __restrict__ bn_b, const float* __restrict__ bn_m,
    const float* __restrict__ bn_v, _Float16* __restrict__ qkvw,
    _Float16* __restrict__ projw, float* __restrict__ biasG,
    float* __restrict__ bnsc, float* __restrict__ bnsh,
    float* __restrict__ qkvbW) {
  const float QSCALE = 0.17677669529663687f;  // hd^-0.5, hd=32
  const int NQ = 3 * 288 * 96;    // 82944
  const int NP = 3 * 96 * 96;     // 27648
  const int NB = 3 * 3 * 64 * 64; // 36864  (br, head, q, k)
  const int NC = 3 * 96;          // 288
  const int NBI = 3 * 288;        // 864
  int i = blockIdx.x * 256 + threadIdx.x;
  if (i < NQ) {
    const int row = (i / 96) % 288;             // output channel within branch
    const float v = qkv_w[i];
    qkvw[i] = (_Float16)(row < 96 ? v * QSCALE : v);
    return;
  }
  i -= NQ;
  if (i < NP) { projw[i] = (_Float16)proj_w[i]; return; }
  i -= NP;
  if (i < NB) {
    const int m = i & 63;
    const int n = (i >> 6) & 63;
    const int h = (i >> 12) % 3;
    const int br = i / 12288;
    biasG[i] = rpb_table[(br * 225 + rpb_index[n * 64 + m]) * 3 + h];
    return;
  }
  i -= NB;
  if (i < NC) {
    const float sc = bn_g[i] * rsqrtf(bn_v[i] + 1e-5f);
    bnsc[i] = sc;
    bnsh[i] = bn_b[i] - bn_m[i] * sc;
    return;
  }
  i -= NC;
  if (i < NBI) {
    const int rc = i % 288;
    const float v = qkv_b[i];
    qkvbW[i] = (rc < 96) ? v * QSCALE : v;
  }
}

__global__ __launch_bounds__(256) void swin3d_zero_kernel(float4* __restrict__ p,
                                                          int n4) {
  int i = blockIdx.x * blockDim.x + threadIdx.x;
  const int stride = gridDim.x * blockDim.x;
  const float4 z = make_float4(0.f, 0.f, 0.f, 0.f);
  for (; i < n4; i += stride) p[i] = z;
}

// ---------------------------------------------------------------------------
// Main fused kernel: one block per (branch, window).
// ---------------------------------------------------------------------------
__global__ __launch_bounds__(256) void swin3d_attn_kernel(
    const float* __restrict__ x, const float* __restrict__ qkvbW,
    const float* __restrict__ proj_b, const _Float16* __restrict__ qkvw,
    const _Float16* __restrict__ projw, const float* __restrict__ biasG,
    const float* __restrict__ bnsc, const float* __restrict__ bnsh,
    float* __restrict__ out) {
  __shared__ __align__(16) _Float16 sXO[N_TOK * XO_STR];       // X, later O
  __shared__ __align__(16) _Float16 sQK[2 * N_TOK * QK_STR];   // Q then K
  __shared__ __align__(16) _Float16 sVt[DIM_ * VT_STR];        // V transposed
  __shared__ __align__(16) float    sS[N_TOK * S_STR];
  __shared__ __align__(16) _Float16 sP[N_TOK * P_STR];

  const int tid = threadIdx.x;
  const int lane = tid & 31;
  const int wave = __builtin_amdgcn_readfirstlane(tid >> 5);

  const int gb = blockIdx.x;
  const int br = gb >> 13;        // 8192 windows per branch
  const int win = gb & 8191;
  const int img = win >> 6;
  const int wy = (win >> 3) & 7;
  const int wx = win & 7;
  const int b = img >> 6;
  const int s = img & 63;         // branch-specific "sequence" coordinate

  // warm WGP$ with the per-branch weight blocks (global_prefetch)
  if (tid < 8) {
    __builtin_prefetch(qkvw + br * (288 * 96) + tid * 10368, 0, 1);
    __builtin_prefetch(projw + br * (96 * 96) + tid * 1152, 0, 1);
    __builtin_prefetch(biasG + br * 12288 + tid * 1536, 0, 1);
  }

  // ---- gather window tokens into LDS (f16), coalesced over tokens ----
  {
    const int tk = tid & 63;
    const int u = wy * 8 + (tk >> 3);
    const int vv = wx * 8 + (tk & 7);
    int sp;
    if (br == 0)      sp = s * 4096 + u * 64 + vv;   // (b d) h w
    else if (br == 1) sp = vv * 4096 + u * 64 + s;   // (b w) h d
    else              sp = u * 4096 + s * 64 + vv;   // (b h) d w
    const int base = b * (96 * 262144) + sp;
    for (int c = (tid >> 6); c < DIM_; c += 4)
      sXO[tk * XO_STR + c] = (_Float16)x[base + c * 262144];
  }
  __syncthreads();

  // ---- QKV GEMM: (64x96) @ W^T(96x288) + b ; 72 tiles / 8 waves ----
  for (int i = 0; i < 9; ++i) {
    const int t = wave + i * 8;             // wave-uniform scalar
    const int mt = t & 3;
    const int nt = t >> 2;                  // 0..17
    const float bias = qkvbW[br * 288 + nt * 16 + (lane & 15)];
    v8f acc;
#pragma unroll
    for (int j = 0; j < 8; ++j) acc[j] = bias;
    const _Float16* wb = qkvw + br * (288 * 96);
#pragma unroll
    for (int k0 = 0; k0 < 96; k0 += 32) {
      v16h a = frag_a(sXO, XO_STR, mt * 16, k0, lane);
      v16h bm = frag_bt(wb, 96, nt * 16, k0, lane);
      WMMA(acc, a, bm);
    }
    // 0=Q 1=K 2=V, forced wave-uniform scalar -> plain scalar branch
    const int sel = __builtin_amdgcn_readfirstlane((unsigned)nt / 6u);
    const int cc = (nt - sel * 6) * 16 + (lane & 15);
    if (sel < 2) {
      _Float16* dst = sQK + sel * (N_TOK * QK_STR);
#pragma unroll
      for (int j = 0; j < 8; ++j) {
        const int row = mt * 16 + j + ((lane >> 4) << 3);
        dst[row * QK_STR + cc] = (_Float16)acc[j];
      }
    } else {
#pragma unroll
      for (int j = 0; j < 8; ++j) {
        const int row = mt * 16 + j + ((lane >> 4) << 3);
        sVt[cc * VT_STR + row] = (_Float16)acc[j];   // contiguous -> b128
      }
    }
  }
  __syncthreads();

  // ---- per-head attention ----
  for (int h = 0; h < 3; ++h) {
    // S = Qh @ Kh^T + rpb_bias : 16 tiles, 2 per wave, K-major K acts as B^T
#pragma unroll
    for (int tt = 0; tt < 2; ++tt) {
      const int t = wave * 2 + tt;
      const int mt = t >> 2, nt = t & 3;
      const float* bp =
          biasG + (((br * 3 + h) * 64) + mt * 16) * 64 + nt * 16 + (lane & 15);
      v8f acc;
#pragma unroll
      for (int j = 0; j < 8; ++j) acc[j] = bp[(j + ((lane >> 4) << 3)) * 64];
      v16h a = frag_a(sQK, QK_STR, mt * 16, h * HD_, lane);
      v16h bm = frag_bt(sQK + N_TOK * QK_STR, QK_STR, nt * 16, h * HD_, lane);
      WMMA(acc, a, bm);
#pragma unroll
      for (int j = 0; j < 8; ++j)
        sS[(mt * 16 + j + ((lane >> 4) << 3)) * S_STR + nt * 16 + (lane & 15)] =
            acc[j];
    }
    __syncthreads();

    // row softmax: 4 lanes per row (lanes 4r..4r+3 within one wave32),
    // 16 values in registers, cross-lane combine via shuffles.
    {
      const int row = tid >> 2;
      const int sub = tid & 3;
      float v[16];
      float mx = -3.0e38f;
#pragma unroll
      for (int i = 0; i < 16; ++i) {
        v[i] = sS[row * S_STR + sub * 16 + i];
        mx = fmaxf(mx, v[i]);
      }
      mx = fmaxf(mx, __shfl_xor(mx, 1, 32));
      mx = fmaxf(mx, __shfl_xor(mx, 2, 32));
      float sum = 0.f;
#pragma unroll
      for (int i = 0; i < 16; ++i) {
        v[i] = __expf(v[i] - mx);
        sum += v[i];
      }
      sum += __shfl_xor(sum, 1, 32);
      sum += __shfl_xor(sum, 2, 32);
      const float inv = 1.0f / sum;
#pragma unroll
      for (int i = 0; i < 16; ++i)
        sP[row * P_STR + sub * 16 + i] = (_Float16)(v[i] * inv);
    }
    __syncthreads();

    // O_h = P(64x64) @ V_h(64x32): 8 tiles, 1 per wave, 2 k-steps
    {
      const int mt = wave >> 1, nt = wave & 1;
      v8f acc;
#pragma unroll
      for (int j = 0; j < 8; ++j) acc[j] = 0.f;
#pragma unroll
      for (int k0 = 0; k0 < 64; k0 += 32) {
        v16h a = frag_a(sP, P_STR, mt * 16, k0, lane);
        v16h bm = frag_bt(sVt + (h * HD_) * VT_STR, VT_STR, nt * 16, k0, lane);
        WMMA(acc, a, bm);
      }
#pragma unroll
      for (int j = 0; j < 8; ++j)
        sXO[(mt * 16 + j + ((lane >> 4) << 3)) * XO_STR + h * HD_ + nt * 16 +
            (lane & 15)] = (_Float16)acc[j];
    }
    __syncthreads();
  }

  // ---- projection + BN + LeakyReLU + accumulate into out ----
  for (int i = 0; i < 3; ++i) {
    const int t = wave + i * 8;
    const int mt = __builtin_amdgcn_readfirstlane((unsigned)t / 6u);
    const int nt = t - mt * 6;
    const int colc = nt * 16 + (lane & 15);
    const float bias = proj_b[br * 96 + colc];
    v8f acc;
#pragma unroll
    for (int j = 0; j < 8; ++j) acc[j] = bias;
    const _Float16* wb = projw + br * (96 * 96);
#pragma unroll
    for (int k0 = 0; k0 < 96; k0 += 32) {
      v16h a = frag_a(sXO, XO_STR, mt * 16, k0, lane);
      v16h bm = frag_bt(wb, 96, nt * 16, k0, lane);
      WMMA(acc, a, bm);
    }
    const float sc = bnsc[br * 96 + colc];
    const float sh = bnsh[br * 96 + colc];
    const int cbase = b * (96 * 262144) + colc * 262144;
#pragma unroll
    for (int j = 0; j < 8; ++j) {
      const int tok = mt * 16 + j + ((lane >> 4) << 3);
      const int u = wy * 8 + (tok >> 3);
      const int vv = wx * 8 + (tok & 7);
      int sp;
      if (br == 0)      sp = s * 4096 + u * 64 + vv;
      else if (br == 1) sp = vv * 4096 + u * 64 + s;
      else              sp = u * 4096 + s * 64 + vv;
      float y = acc[j] * sc + sh;
      y = (y >= 0.f) ? y : 0.01f * y;
      atomicAdd(out + cbase + sp, y);
    }
  }
}

// ---------------------------------------------------------------------------
extern "C" void kernel_launch(void* const* d_in, const int* in_sizes, int n_in,
                              void* d_out, int out_size, void* d_ws,
                              size_t ws_size, hipStream_t stream) {
  (void)in_sizes; (void)n_in; (void)ws_size;
  const float* x         = (const float*)d_in[0];
  const float* qkv_w     = (const float*)d_in[1];
  const float* qkv_b     = (const float*)d_in[2];
  const float* proj_w    = (const float*)d_in[3];
  const float* proj_b    = (const float*)d_in[4];
  const float* rpb_table = (const float*)d_in[5];
  const float* bn_g      = (const float*)d_in[6];
  const float* bn_b      = (const float*)d_in[7];
  const float* bn_m      = (const float*)d_in[8];
  const float* bn_v      = (const float*)d_in[9];
  const int*   rpb_index = (const int*)d_in[10];
  float* out = (float*)d_out;

  char* ws = (char*)d_ws;
  _Float16* qkvw  = (_Float16*)(ws);            // 3*288*96 f16 = 165888 B
  _Float16* projw = (_Float16*)(ws + 165888);   // 3*96*96  f16 =  55296 B
  float*    biasG = (float*)(ws + 221184);      // 3*3*64*64 f32 = 147456 B
  float*    bnsc  = (float*)(ws + 368640);      // 288 f32
  float*    bnsh  = (float*)(ws + 369792);      // 288 f32
  float*    qkvbW = (float*)(ws + 370944);      // 3*288 f32 (scaled)

  // zero the accumulation target
  const int n4 = out_size / 4;
  swin3d_zero_kernel<<<8192, 256, 0, stream>>>((float4*)out, n4);

  // prep: 82944 + 27648 + 36864 + 288 + 864 = 148608 items
  swin3d_prep_kernel<<<(148608 + 255) / 256, 256, 0, stream>>>(
      qkv_w, qkv_b, proj_w, rpb_table, rpb_index, bn_g, bn_b, bn_m, bn_v, qkvw,
      projw, biasG, bnsc, bnsh, qkvbW);

  // 3 branches * 8192 windows
  swin3d_attn_kernel<<<3 * 8192, 256, 0, stream>>>(
      x, qkvbW, proj_b, qkvw, projw, biasG, bnsc, bnsh, out);
}
